// STRNN_64218351010370
// MI455X (gfx1250) — matrix operand order repaired and checked by reading, not compiled, not run on hardware
//
#include <hip/hip_runtime.h>
#include <hip/hip_bf16.h>

// ---------------------------------------------------------------------------
// STRNN on MI455X (gfx1250, wave32, WMMA + async global->LDS).
// Pipeline:
//   K0: f32->f16 convert; weights stored transposed (column-major) so WMMA
//       B-fragments are contiguous 32B per lane.
//   K1: XT[t] = X @ time_weights[t], 97 GEMMs via v_wmma_f32_16x16x32_f16.
//   K2: per (b,i): LDS bucket accumulate A[d,0:64] += XT[t_idx][b,s] over
//       valid s (ds_add_f32 atomics), emit A_flat row (6208 f16).
//   K3: XC = A_flat(2048x6208) @ W_flat(6208x128) via WMMA; the block-shared
//       16-row A chunk is double-buffered into LDS with
//       global_load_async_to_lds_b128 / s_wait_asynccnt (ASYNCcnt pipe).
//   K4: sequential sigmoid recurrence, one block per batch, W in LDS.
// ---------------------------------------------------------------------------

typedef __attribute__((ext_vector_type(16))) _Float16 v16h;
typedef __attribute__((ext_vector_type(8)))  _Float16 v8h;
typedef __attribute__((ext_vector_type(8)))  float    v8f;

#define Bq 16
#define Sq 128
#define Dq 64
#define Kq 64
#define Hq 128
#define NS 97          // NSLOT + 1
#define N1 2048        // B*S rows
#define KF 6208        // NS*Kq flattened GEMM-K

// workspace byte offsets (all 256B aligned)
#define OFF_XH   ((size_t)0)          // 2048*64 f16      = 262144
#define OFF_TWT  ((size_t)262144)     // 97*64*64 f16     = 794624
#define OFF_DWT  ((size_t)1056768)    // 97*128*64 f16    = 1589248
#define OFF_XT   ((size_t)2646016)    // 97*2048*64 f16   = 25427968
#define OFF_AF   ((size_t)28073984)   // 2048*6208 f16    = 25427968
#define OFF_XC   ((size_t)53501952)   // 2048*128 f32     = 1048576

// ---- K0: convert + transpose weights ---------------------------------------
__global__ void k_convert(const float* __restrict__ x,
                          const float* __restrict__ tw,
                          const float* __restrict__ dw,
                          _Float16* __restrict__ Xh,
                          _Float16* __restrict__ TWT,
                          _Float16* __restrict__ DWT) {
  const int stride = gridDim.x * blockDim.x;
  const int tid0 = blockIdx.x * blockDim.x + threadIdx.x;
  // Xh: straight row-major copy (rows n=b*S+s, cols d)
  for (int i = tid0; i < N1 * Dq; i += stride)
    Xh[i] = (_Float16)x[i];
  // TWT[t][n][d] = time_weights[t][d][n]   (B operand of K1, column-major)
  for (int i = tid0; i < NS * Kq * Dq; i += stride) {
    int t = i >> 12, r = i & 4095, n = r >> 6, dd = r & 63;
    TWT[i] = (_Float16)tw[(t << 12) + (dd << 6) + n];
  }
  // DWT[d][h][k] = dist_weights[d][k][h]   (B operand of K3, column-major)
  for (int i = tid0; i < NS * Hq * Kq; i += stride) {
    int d = i >> 13, r = i & 8191, h = r >> 6, k = r & 63;
    DWT[i] = (_Float16)dw[(d << 13) + (k << 7) + h];
  }
}

// A fragment (16x32 f16) in contiguous k order (B uses the same):
// lane = hi*16+m ; element e -> K = k0 + 16*hi + e
__device__ __forceinline__ v16h load_afrag(const _Float16* __restrict__ row,
                                           int k0, int hi) {
  v8h lo = *(const v8h*)(row + k0 + 16 * hi);
  v8h hh = *(const v8h*)(row + k0 + 16 * hi + 8);
  v16h a;
#pragma unroll
  for (int e = 0; e < 8; ++e) { a[e] = lo[e]; a[e + 8] = hh[e]; }
  return a;
}

// ---- K1: XT[t] = X @ TW[t]  (97 GEMMs, M=2048 N=64 K=64) -------------------
__global__ void k_xt_gemm(const _Float16* __restrict__ Xh,
                          const _Float16* __restrict__ TWT,
                          _Float16* __restrict__ XT) {
  const int wave = blockIdx.x * (blockDim.x >> 5) + (threadIdx.x >> 5);
  const int lane = threadIdx.x & 31;
  const int t  = wave >> 9;          // 97
  const int rm = wave & 511;
  const int mt = rm >> 2;            // 128 M-tiles
  const int nt = rm & 3;             // 4  N-tiles
  const int m = lane & 15, hi = lane >> 4;

  const _Float16* Arow = Xh + (size_t)(mt * 16 + m) * Dq;
  const _Float16* Bcol = TWT + ((size_t)t * Kq + nt * 16 + m) * Dq;
  v8f acc = {};
#pragma unroll
  for (int k0 = 0; k0 < Dq; k0 += 32) {
    v16h a = load_afrag(Arow, k0, hi);
    v16h b = *(const v16h*)(Bcol + k0 + 16 * hi);  // contiguous K for column n
    acc = __builtin_amdgcn_wmma_f32_16x16x32_f16(false, a, false, b,
                                                 (short)0, acc, false, false);
  }
  // C layout: VGPR r -> row m=r+8*hi, col = lane&15
#pragma unroll
  for (int r = 0; r < 8; ++r)
    XT[((size_t)t * N1 + mt * 16 + r + 8 * hi) * Kq + nt * 16 + m] =
        (_Float16)acc[r];
}

// ---- K2: slot-bucket accumulation in LDS, one block per (b,i) --------------
__global__ void k_bucket(const int* __restrict__ tc,
                         const int* __restrict__ dc,
                         const unsigned char* __restrict__ mk,
                         const _Float16* __restrict__ XT,
                         _Float16* __restrict__ Af) {
  __shared__ float Ab[KF];  // 97 buckets x 64 f32 = 24.8 KB of the 320 KB LDS
  const int n2 = blockIdx.x;           // b*S + i
  const int b = n2 >> 7, i = n2 & 127;
  const int tid = threadIdx.x;
  for (int j = tid; j < KF; j += 256) Ab[j] = 0.0f;
  __syncthreads();

  const int wid = tid >> 5, lane = tid & 31;
  const int* trow = tc + (size_t)n2 * Sq;
  const int* drow = dc + (size_t)n2 * Sq;
  const unsigned char* mrow = mk + (size_t)n2 * Sq;
  for (int s = wid; s <= i; s += 8) {        // causal mask s <= i
    if (!mrow[s]) continue;
    const int t = trow[s], d = drow[s];
    const _Float16* xr = XT + ((size_t)t * N1 + b * Sq + s) * Kq;
    atomicAdd(&Ab[(d << 6) + lane],       (float)xr[lane]);        // ds_add_f32
    atomicAdd(&Ab[(d << 6) + lane + 32],  (float)xr[lane + 32]);
  }
  __syncthreads();
  _Float16* out = Af + (size_t)n2 * KF;
  for (int j = tid; j < KF; j += 256) out[j] = (_Float16)Ab[j];
}

// ---- K3: XC = A_flat(2048x6208) @ W_flat(6208x128) -------------------------
// One block per M-tile; 8 waves cover the 8 N-tiles, so the 16-row A chunk is
// shared. It is double-buffered into LDS with async global->LDS loads issued
// by waves 0-1 and overlapped with the WMMA of the current chunk.
#define AB_ROW 40                 // 32 f16 used + 8 pad (20-word bank stride)
#define AB_BUF (16 * AB_ROW)      // f16 per buffer

__global__ void __launch_bounds__(256) k_xc_gemm(
    const _Float16* __restrict__ Af,
    const _Float16* __restrict__ DWT,
    float* __restrict__ XC) {
  __shared__ _Float16 Ab[2 * AB_BUF];   // 2 x 1.25 KB
  const int tid = threadIdx.x;
  const int wv = tid >> 5, lane = tid & 31;
  const int mt = blockIdx.x;            // 128 M-tiles
  const int nt = wv;                    // 8 N-tiles (one per wave)
  const int m = lane & 15, hi = lane >> 4;

  // async staging: threads 0..63 (waves 0-1) each fetch one b128 per chunk
  const int arow = tid >> 2;            // 0..15 source row in the M-tile
  const int aq   = tid & 3;             // 0..3  16B quarter of the 64B chunk
  const unsigned lds0 = (unsigned)(size_t)&Ab[0];
  const _Float16* asrc = Af + (size_t)(mt * 16 + arow) * KF + aq * 8;

#define ISSUE_ASYNC(buf, kk)                                                 \
  {                                                                          \
    unsigned l = lds0 + (unsigned)((buf) * (AB_BUF * 2) +                    \
                                   arow * (AB_ROW * 2) + aq * 16);           \
    unsigned long long g = (unsigned long long)(size_t)(asrc + (kk));        \
    asm volatile("global_load_async_to_lds_b128 %0, %1, off"                 \
                 :: "v"(l), "v"(g) : "memory");                              \
  }

  if (tid < 64) ISSUE_ASYNC(0, 0);

  const _Float16* Bbase = DWT + (size_t)(nt * 16 + m) * Kq;
  v8f acc = {};
  for (int kk = 0; kk < KF; kk += 32) {
    const int c = kk >> 5;
    if (tid < 64)                       // wave-uniform branch (waves 0-1)
      asm volatile("s_wait_asynccnt 0x0" ::: "memory");
    __syncthreads();                    // chunk c visible to all 8 waves
    if (tid < 64 && kk + 32 < KF) ISSUE_ASYNC((c + 1) & 1, kk + 32);

    const int d0 = kk >> 6, ko = kk & 63;
    const _Float16* arow_lds = &Ab[(c & 1) * AB_BUF + m * AB_ROW];
    v8h lo = *(const v8h*)(arow_lds + 16 * hi);       // ds_load_b128
    v8h hh = *(const v8h*)(arow_lds + 16 * hi + 8);   // ds_load_b128
    v16h a;
#pragma unroll
    for (int e = 0; e < 8; ++e) { a[e] = lo[e]; a[e + 8] = hh[e]; }

    const _Float16* bp = Bbase + (size_t)d0 * (Hq * Kq) + ko + 16 * hi;
    if (kk + 128 < KF)
      __builtin_prefetch(bp + 128, 0, 1);             // global_prefetch_b8
    v16h b = *(const v16h*)bp;
    acc = __builtin_amdgcn_wmma_f32_16x16x32_f16(false, a, false, b,
                                                 (short)0, acc, false, false);
    __syncthreads();                    // all reads of buffer c done
  }
#pragma unroll
  for (int r = 0; r < 8; ++r)
    XC[((size_t)(mt * 16 + r + 8 * hi)) * Hq + nt * 16 + m] = acc[r];
#undef ISSUE_ASYNC
}

// ---- K4: sequential sigmoid recurrence, one block per batch ----------------
__global__ void k_rnn(const float* __restrict__ XC,
                      const float* __restrict__ HW,
                      float* __restrict__ out) {
  __shared__ float Wl[Hq * Hq];  // 64 KB hidden matrix in LDS
  __shared__ float hb[Hq];
  const int b = blockIdx.x, tid = threadIdx.x;
  for (int j = tid; j < Hq * Hq; j += Hq) Wl[j] = HW[j];
  hb[tid] = 0.0f;
  __syncthreads();
  for (int i = 0; i < Sq; ++i) {
    float acc = XC[((size_t)b * Sq + i) * Hq + tid];
#pragma unroll 8
    for (int j = 0; j < Hq; ++j) acc += hb[j] * Wl[j * Hq + tid];
    const float sg = 1.0f / (1.0f + __expf(-acc));
    out[((size_t)b * Sq + i) * Hq + tid] = sg;   // outs[b,i,h]
    __syncthreads();
    hb[tid] = sg;
    __syncthreads();
  }
  out[(size_t)Bq * Sq * Hq + b * Hq + tid] = hb[tid];  // hT[0,b,h]
}

// ---------------------------------------------------------------------------
extern "C" void kernel_launch(void* const* d_in, const int* in_sizes, int n_in,
                              void* d_out, int out_size, void* d_ws,
                              size_t ws_size, hipStream_t stream) {
  (void)in_sizes; (void)n_in; (void)out_size; (void)ws_size;
  const float* x          = (const float*)d_in[0];
  const int* tc           = (const int*)d_in[1];
  const int* dc           = (const int*)d_in[2];
  const unsigned char* mk = (const unsigned char*)d_in[3];  // bool mask
  const float* tw         = (const float*)d_in[4];
  const float* dw         = (const float*)d_in[5];
  const float* hw         = (const float*)d_in[6];
  float* out              = (float*)d_out;

  char* ws = (char*)d_ws;
  _Float16* Xh  = (_Float16*)(ws + OFF_XH);
  _Float16* TWT = (_Float16*)(ws + OFF_TWT);
  _Float16* DWT = (_Float16*)(ws + OFF_DWT);
  _Float16* XT  = (_Float16*)(ws + OFF_XT);
  _Float16* Af  = (_Float16*)(ws + OFF_AF);
  float*    XC  = (float*)   (ws + OFF_XC);

  hipLaunchKernelGGL(k_convert, dim3(1024), dim3(256), 0, stream,
                     x, tw, dw, Xh, TWT, DWT);
  hipLaunchKernelGGL(k_xt_gemm, dim3(6208), dim3(256), 0, stream,   // 97*512 waves
                     Xh, TWT, XT);
  hipLaunchKernelGGL(k_bucket, dim3(2048), dim3(256), 0, stream,
                     tc, dc, mk, XT, Af);
  hipLaunchKernelGGL(k_xc_gemm, dim3(128), dim3(256), 0, stream,    // 1024 waves
                     Af, DWT, XC);
  hipLaunchKernelGGL(k_rnn, dim3(16), dim3(128), 0, stream, XC, hw, out);
}
